// EnhancedVectorQuantizer_33904471835540
// MI455X (gfx1250) — matrix operand order repaired and checked by reading, not compiled, not run on hardware
//
#include <hip/hip_runtime.h>

#define DLAT   256
#define KCODE  8192
#define NROWS  32768
#define MTILE  128
#define KCHUNK 64
#define LDSPITCH 264   // 256 + 8 pad (keeps 16B row alignment, breaks bank conflicts)

typedef __attribute__((ext_vector_type(16))) __bf16 v16bf;
typedef __attribute__((ext_vector_type(8)))  float  v8f;

union BFrag {
    v16bf v;
    uint4 q[2];
    __bf16 e[16];
    unsigned short u[16];
};

union BF1 { __bf16 b; unsigned short u; };

// CDNA5 async global->LDS copy (tracked by ASYNCcnt, bypasses VGPRs & ds_store).
__device__ __forceinline__ void async_b128(unsigned lds_off, unsigned goff, const void* gbase)
{
    asm volatile("global_load_async_to_lds_b128 %0, %1, %2"
                 :: "v"(lds_off), "v"(goff), "s"(gbase)
                 : "memory");
}

__device__ __forceinline__ void wait_asynccnt0()
{
    asm volatile("s_wait_asynccnt 0x0" ::: "memory");
}

// ---------------------------------------------------------------------------
// Kernel 1: split weight into bf16 hi/lo, compute w_sq[k]. One block per code.
// ---------------------------------------------------------------------------
__global__ __launch_bounds__(256)
void vq_prep_kernel(const float* __restrict__ weight,
                    unsigned short* __restrict__ whi,
                    unsigned short* __restrict__ wlo,
                    float* __restrict__ wsq)
{
    __shared__ float red[256];
    const int k = blockIdx.x;
    const int d = threadIdx.x;
    const float w = weight[(size_t)k * DLAT + d];

    BF1 hb; hb.b = (__bf16)w;
    const float hif = (float)hb.b;
    BF1 lb; lb.b = (__bf16)(w - hif);
    whi[(size_t)k * DLAT + d] = hb.u;
    wlo[(size_t)k * DLAT + d] = lb.u;

    red[d] = w * w;
    __syncthreads();
    for (int s = 128; s > 0; s >>= 1) {
        if (d < s) red[d] += red[d + s];
        __syncthreads();
    }
    if (d == 0) wsq[k] = red[0];
}

// ---------------------------------------------------------------------------
// Kernel 2: fused distance GEMM + argmin via bf16 hi/lo split WMMA.
// Block = 128 rows (8 waves x 16-row strips), loops over all K codes.
// dist(n,k) = wsq[k] - 2 * dot(x[n], w[k])   (x_sq omitted: row-constant)
// Codebook chunks are streamed with double-buffered async global->LDS loads.
// ---------------------------------------------------------------------------
__global__ __launch_bounds__(256)
void vq_argmin_kernel(const float* __restrict__ x,
                      const unsigned short* __restrict__ whi,
                      const unsigned short* __restrict__ wlo,
                      const float* __restrict__ wsq,
                      int* __restrict__ idx_out)
{
    __shared__ __align__(16) unsigned short s_whi[2][KCHUNK][LDSPITCH];
    __shared__ __align__(16) unsigned short s_wlo[2][KCHUNK][LDSPITCH];

    const int tid  = threadIdx.x;
    const int lane = tid & 31;
    const int wave = tid >> 5;
    const int h    = lane >> 4;   // lane half (ISA fragment layout)
    const int col  = lane & 15;   // fixed C-matrix column per lane
    const int m0   = blockIdx.x * MTILE;

    // Async staging of one 64-code chunk (hi+lo) into LDS buffer `buf`.
    auto stage = [&](int buf, int kc) {
        const unsigned short* gh = whi + (size_t)kc * DLAT;
        const unsigned short* gl = wlo + (size_t)kc * DLAT;
#pragma unroll
        for (int j = 0; j < 8; ++j) {
            const int id  = tid + 256 * j;   // 0..2047 x 16B (32 per code row)
            const int r   = id >> 5;
            const int c16 = id & 31;
            const unsigned dsth = (unsigned)(uintptr_t)&s_whi[buf][r][c16 * 8];
            const unsigned dstl = (unsigned)(uintptr_t)&s_wlo[buf][r][c16 * 8];
            async_b128(dsth, (unsigned)(id * 16), gh);
            async_b128(dstl, (unsigned)(id * 16), gl);
        }
    };

    // ---- Build resident A fragments (hi/lo) for this wave's 16 rows ----
    // 16-bit A 16x32 layout: elems 0..15 = k[32c+16h .. +15] (contiguous, matches B)
    const int arow = m0 + wave * 16 + col;      // A-matrix row M = lane&15
    const float4* x4 = (const float4*)(x + (size_t)arow * DLAT);

    // Kick off the first chunk while we convert A fragments.
    stage(0, 0);

    v16bf a_hi[8], a_lo[8];
#pragma unroll
    for (int c = 0; c < 8; ++c) {
        const int d0 = 32 * c + 16 * h;
        const int d1 = 32 * c + 16 * h + 8;
        float fv[16];
        float4 p;
        p = x4[(d0 >> 2) + 0]; fv[0] = p.x; fv[1] = p.y; fv[2]  = p.z; fv[3]  = p.w;
        p = x4[(d0 >> 2) + 1]; fv[4] = p.x; fv[5] = p.y; fv[6]  = p.z; fv[7]  = p.w;
        p = x4[(d1 >> 2) + 0]; fv[8] = p.x; fv[9] = p.y; fv[10] = p.z; fv[11] = p.w;
        p = x4[(d1 >> 2) + 1]; fv[12]= p.x; fv[13]= p.y; fv[14] = p.z; fv[15] = p.w;
        BFrag fh, fl;
#pragma unroll
        for (int i = 0; i < 16; ++i) {
            __bf16 hi = (__bf16)fv[i];
            fh.e[i] = hi;
            fl.e[i] = (__bf16)(fv[i] - (float)hi);
        }
        a_hi[c] = fh.v;
        a_lo[c] = fl.v;
    }

    float minv[8];
    int   mini[8];
#pragma unroll
    for (int v = 0; v < 8; ++v) { minv[v] = 3.4e38f; mini[v] = 0x7fffffff; }

    int buf = 0;
    for (int kc = 0; kc < KCODE; kc += KCHUNK) {
        wait_asynccnt0();      // my wave's staged data has landed in LDS
        __syncthreads();       // all waves staged + all done reading other buffer
        if (kc + KCHUNK < KCODE) stage(buf ^ 1, kc + KCHUNK);   // prefetch next chunk

#pragma unroll
        for (int s = 0; s < 4; ++s) {
            const unsigned short* bh_row = &s_whi[buf][16 * s + col][0];
            const unsigned short* bl_row = &s_wlo[buf][16 * s + col][0];

            // Software-pipelined B fragments (one depth-chunk ahead).
            // 16-bit B 32x16 layout: elems 0..15 = k[32c+16h .. +15] (contiguous)
            BFrag bh, bl;
            bh.q[0] = *(const uint4*)&bh_row[16 * h];
            bh.q[1] = *(const uint4*)&bh_row[16 * h + 8];
            bl.q[0] = *(const uint4*)&bl_row[16 * h];
            bl.q[1] = *(const uint4*)&bl_row[16 * h + 8];

            v8f acc = {0.f, 0.f, 0.f, 0.f, 0.f, 0.f, 0.f, 0.f};
#pragma unroll
            for (int c = 0; c < 8; ++c) {
                BFrag nbh, nbl;
                if (c < 7) {
                    const int off = 32 * (c + 1) + 16 * h;
                    nbh.q[0] = *(const uint4*)&bh_row[off];
                    nbh.q[1] = *(const uint4*)&bh_row[off + 8];
                    nbl.q[0] = *(const uint4*)&bl_row[off];
                    nbl.q[1] = *(const uint4*)&bl_row[off + 8];
                }
                // dot = hi*hi + lo*hi + hi*lo  (lo*lo negligible)
                acc = __builtin_amdgcn_wmma_f32_16x16x32_bf16(false, a_hi[c], false, bh.v,
                                                              (short)0, acc, false, false);
                acc = __builtin_amdgcn_wmma_f32_16x16x32_bf16(false, a_lo[c], false, bh.v,
                                                              (short)0, acc, false, false);
                acc = __builtin_amdgcn_wmma_f32_16x16x32_bf16(false, a_hi[c], false, bl.v,
                                                              (short)0, acc, false, false);
                if (c < 7) { bh = nbh; bl = nbl; }
            }

            const int code = kc + 16 * s + col;
            const float wsqv = wsq[code];
#pragma unroll
            for (int v = 0; v < 8; ++v) {
                const float dist = fmaf(-2.0f, acc[v], wsqv);
                if (dist < minv[v] || (dist == minv[v] && code < mini[v])) {
                    minv[v] = dist;
                    mini[v] = code;
                }
            }
        }
        buf ^= 1;
    }

    // Cross-lane argmin over the 16 columns held by each 16-lane group.
#pragma unroll
    for (int v = 0; v < 8; ++v) {
        float mv = minv[v];
        int   mi = mini[v];
#pragma unroll
        for (int off = 1; off < 16; off <<= 1) {
            const float ov = __shfl_xor(mv, off, 32);
            const int   oi = __shfl_xor(mi, off, 32);
            if (ov < mv || (ov == mv && oi < mi)) { mv = ov; mi = oi; }
        }
        if (col == 0) {
            const int r = m0 + wave * 16 + 8 * h + v;   // C row M = v + 8h
            idx_out[r] = mi;
        }
    }
}

// ---------------------------------------------------------------------------
// Kernel 3: gather quantized rows, write quantized_st (== quantized values),
// emit indices as float, and deterministic per-row squared-error partials.
// ---------------------------------------------------------------------------
__global__ __launch_bounds__(256)
void vq_quant_loss_kernel(const float* __restrict__ x,
                          const float* __restrict__ weight,
                          const int* __restrict__ idx_ws,
                          float* __restrict__ out_q,
                          float* __restrict__ out_idx,
                          float* __restrict__ partial)
{
    __shared__ float red[256];
    const int n = blockIdx.x;
    const int d = threadIdx.x;
    const int idx = idx_ws[n];
    const float q  = weight[(size_t)idx * DLAT + d];
    const float xv = x[(size_t)n * DLAT + d];
    out_q[(size_t)n * DLAT + d] = q;   // x + sg(q - x) == q numerically
    const float diff = xv - q;
    red[d] = diff * diff;
    __syncthreads();
    for (int s = 128; s > 0; s >>= 1) {
        if (d < s) red[d] += red[d + s];
        __syncthreads();
    }
    if (d == 0) {
        partial[n] = red[0];
        out_idx[n] = (float)idx;
    }
}

// ---------------------------------------------------------------------------
// Kernel 4: deterministic final reduction -> loss scalars.
// ---------------------------------------------------------------------------
__global__ __launch_bounds__(256)
void vq_finalize_kernel(const float* __restrict__ partial,
                        float* __restrict__ out_sc)
{
    __shared__ float red[256];
    float s = 0.0f;
    for (int i = threadIdx.x; i < NROWS; i += 256) s += partial[i];
    red[threadIdx.x] = s;
    __syncthreads();
    for (int t = 128; t > 0; t >>= 1) {
        if (threadIdx.x < t) red[threadIdx.x] += red[threadIdx.x + t];
        __syncthreads();
    }
    if (threadIdx.x == 0) {
        const float msd = red[0] / 8388608.0f;   // N * D
        out_sc[0] = 0.25f * msd;   // commitment_loss
        out_sc[1] = msd;           // codebook_loss
        out_sc[2] = 1.25f * msd;   // total
    }
}

// ---------------------------------------------------------------------------
extern "C" void kernel_launch(void* const* d_in, const int* in_sizes, int n_in,
                              void* d_out, int out_size, void* d_ws, size_t ws_size,
                              hipStream_t stream)
{
    (void)in_sizes; (void)n_in; (void)out_size; (void)ws_size;

    const float* x      = (const float*)d_in[0];   // [8,4096,256]
    const float* weight = (const float*)d_in[1];   // [8192,256]

    char* ws = (char*)d_ws;
    unsigned short* whi = (unsigned short*)(ws);                        // 4 MB
    unsigned short* wlo = (unsigned short*)(ws + (4u << 20));           // 4 MB
    float* wsq  = (float*)(ws + (8u << 20));                            // 32 KB
    int*   idxw = (int*)  (ws + (8u << 20) + (32u << 10));              // 128 KB
    float* part = (float*)(ws + (8u << 20) + (160u << 10));             // 128 KB

    float* out      = (float*)d_out;
    float* out_q    = out;                              // 8388608 f32
    float* out_idx  = out + (size_t)NROWS * DLAT;       // 32768 f32 (indices)
    float* out_sc   = out_idx + NROWS;                  // 3 scalars

    vq_prep_kernel<<<KCODE, 256, 0, stream>>>(weight, whi, wlo, wsq);
    vq_argmin_kernel<<<NROWS / MTILE, 256, 0, stream>>>(x, whi, wlo, wsq, idxw);
    vq_quant_loss_kernel<<<NROWS, 256, 0, stream>>>(x, weight, idxw, out_q, out_idx, part);
    vq_finalize_kernel<<<1, 256, 0, stream>>>(part, out_sc);
}
